// GRU_12292196401828
// MI455X (gfx1250) — matrix operand order repaired and checked by reading, not compiled, not run on hardware
//
#include <hip/hip_runtime.h>
#include <cstdint>

// B=64, T=2048, I=256, H=256 GRU forward for MI455X (gfx1250).
// Phase 1: gi = x @ W_ih^T + biases as a wide bf16-WMMA GEMM. Each block owns
//          one 16-row M tile: x tile loaded from HBM once, converted f32->bf16
//          once into LDS; 8 waves cover all 768 N columns (6 tiles each).
// Phase 2: 4 persistent workgroups (16 batch rows each) run the 2048-step
//          recurrence with W_hh register-stationary and gi async-prefetched
//          into a double-buffered LDS stage (ASYNCcnt-tracked).

typedef __attribute__((ext_vector_type(16))) __bf16 v16bf;
typedef __attribute__((ext_vector_type(8)))  __bf16 v8bf;
typedef __attribute__((ext_vector_type(8)))  float  v8f;

#define B_  64
#define T_  2048
#define I_  256
#define H_  256
#define G3  768   // 3*H

static __device__ __forceinline__ v16bf cat8(v8bf lo, v8bf hi) {
  v16bf r;
#pragma unroll
  for (int i = 0; i < 8; ++i) { r[i] = lo[i]; r[i + 8] = hi[i]; }
  return r;
}

static __device__ __forceinline__ float sigmoid_(float x) {
  return 1.0f / (1.0f + __expf(-x));
}
static __device__ __forceinline__ float tanh_(float x) {
  x = fminf(fmaxf(x, -12.0f), 12.0f);
  float e = __expf(-2.0f * x);
  return (1.0f - e) / (1.0f + e);
}

// ---------------- fp32 -> bf16 weight conversion ----------------
__global__ __launch_bounds__(256) void gru_cvt_w(const float* __restrict__ Wih,
                                                 const float* __restrict__ Whh,
                                                 __bf16* __restrict__ Wih_bf,
                                                 __bf16* __restrict__ Whh_bf) {
  int i = blockIdx.x * 256 + threadIdx.x;  // grid covers 3H*I = 196608
  Wih_bf[i] = (__bf16)Wih[i];
  Whh_bf[i] = (__bf16)Whh[i];
}

// ---------------- Phase 1: gi GEMM (M=B*T, N=3H, K=I) ----------------
// One 16-row M tile per block, staged once in LDS as bf16. 8 waves x 6 N-tiles
// cover all 768 columns; 8 K-steps of v_wmma_f32_16x16x32_bf16 (48 WMMA/wave).
__global__ __launch_bounds__(256) void gru_gi_gemm(const float* __restrict__ x,
                                                   const __bf16* __restrict__ Wih_bf,
                                                   const float* __restrict__ b_ih,
                                                   const float* __restrict__ b_hh,
                                                   float* __restrict__ gi) {
  __shared__ __bf16 xa[16 * I_];             // 8 KB: A tile, row-major [16][256]

  const int tid   = threadIdx.x;
  const int lane  = tid & 31;
  const int wave  = tid >> 5;                // 0..7
  const int mbase = blockIdx.x * 16;         // 8192 M tiles
  const int l16   = lane & 15;
  const int hiH   = lane >> 4;               // 0: lanes 0-15, 1: lanes 16-31

  // Cooperative x tile load + f32->bf16 convert: 1024 float4 chunks / 256 thr.
#pragma unroll
  for (int q = 0; q < 4; ++q) {
    const int c    = q * 256 + tid;          // chunk id, 64 float4 per row
    const int row  = c >> 6;
    const int col4 = (c & 63) * 4;
    float4 f = *(const float4*)(x + (size_t)(mbase + row) * I_ + col4);
    const int o = row * I_ + col4;
    xa[o + 0] = (__bf16)f.x; xa[o + 1] = (__bf16)f.y;
    xa[o + 2] = (__bf16)f.z; xa[o + 3] = (__bf16)f.w;
  }
  __syncthreads();

  v8f acc[6];
#pragma unroll
  for (int t = 0; t < 6; ++t)
#pragma unroll
    for (int r = 0; r < 8; ++r) acc[t][r] = 0.0f;

  const int nb = wave * 96;                  // this wave's 96-column strip

#pragma unroll
  for (int kk = 0; kk < 8; ++kk) {
    const int k0 = kk * 32;
    // A fragment (16x32 bf16) from LDS: lanes 0-15 K 0..15, 16-31 K 16..31
    const int abase = l16 * I_ + k0 + hiH * 16;
    v16bf a = cat8(*(const v8bf*)(xa + abase),
                   *(const v8bf*)(xa + abase + 8));

    const int koffB = k0 + hiH * 16;         // B-frag (32x16): 16 contig K / lane
#pragma unroll
    for (int t = 0; t < 6; ++t) {
      const int n = nb + t * 16 + l16;       // B column n == W_ih row n (contig K)
      const v8bf* wp = (const v8bf*)(Wih_bf + (size_t)n * I_ + koffB);
      v16bf b = cat8(wp[0], wp[1]);
      acc[t] = __builtin_amdgcn_wmma_f32_16x16x32_bf16(
          false, a, false, b, (short)0, acc[t], false, false);
    }
  }

#pragma unroll
  for (int t = 0; t < 6; ++t) {
    const int n    = nb + t * 16 + l16;
    const int gate = n / H_;
    // candidate gate keeps b_hh inside the reset-gated term (added in phase 2)
    const float bias = b_ih[n] + (gate == 1 ? 0.0f : b_hh[n]);
#pragma unroll
    for (int r = 0; r < 8; ++r) {
      const int row = hiH ? (r + 8) : r;     // f32 C/D tile layout
      gi[(size_t)(mbase + row) * G3 + n] = acc[t][r] + bias;
    }
  }
}

// ---------------- Phase 2: persistent recurrence ----------------
// 4 blocks x 512 threads (16 waves). Wave w owns H-columns [16w,16w+16) of all
// 3 gates (3 accumulators). W_hh fragments live in registers for all T steps.
// gi[t] (16x768 f32, 48KB) is double-buffered in LDS via async global->LDS b128.
__global__ __launch_bounds__(512) void gru_rnn(const float* __restrict__ gi,
                                               const float* __restrict__ h0,
                                               const __bf16* __restrict__ Whh_bf,
                                               const float* __restrict__ b_hh,
                                               float* __restrict__ out,
                                               float* __restrict__ hN) {
  extern __shared__ char smem[];
  float*  gi_buf = (float*)smem;                                   // 2*16*768 f32
  __bf16* h_bf   = (__bf16*)(smem + 2 * 16 * G3 * 4);              // 16*256 bf16
  float*  h_f    = (float*)(smem + 2 * 16 * G3 * 4 + 16 * H_ * 2); // 16*256 f32

  const int tid  = threadIdx.x;
  const int lane = tid & 31;
  const int wave = tid >> 5;      // 0..15
  const int bb   = blockIdx.x;    // batch block: rows [16*bb, 16*bb+16)
  const int l16  = lane & 15;
  const int hiH  = lane >> 4;
  const int jc   = wave * 16 + l16;   // this lane's H-column

  // Register-stationary W_hh B-fragments: 3 gates x 8 K-steps (192 VGPRs).
  v16bf Bf[3][8];
#pragma unroll
  for (int g = 0; g < 3; ++g) {
    const __bf16* wr = Whh_bf + (size_t)(g * H_ + jc) * H_;
#pragma unroll
    for (int kk = 0; kk < 8; ++kk) {
      const v8bf* wp = (const v8bf*)(wr + kk * 32 + hiH * 16);
      Bf[g][kk] = cat8(wp[0], wp[1]);
    }
  }
  const float bhh_c = b_hh[H_ + jc];   // candidate-gate hidden bias

  for (int i = tid; i < 16 * H_; i += 512) {
    float v = h0[(size_t)(bb * 16 + (i >> 8)) * H_ + (i & 255)];
    h_f[i]  = v;
    h_bf[i] = (__bf16)v;
  }
  __syncthreads();

  const uint32_t lds0 = (uint32_t)(uintptr_t)gi_buf;           // flat low32 == LDS off
  const uint32_t lds1 = (uint32_t)(uintptr_t)(gi_buf + 16 * G3);

  // 48KB per step = 3072 x 16B transfers = 6 per thread.
  auto prefetch = [&](int t, uint32_t ldsbase) {
#pragma unroll
    for (int q = 0; q < 6; ++q) {
      const int i     = q * 512 + tid;       // 0..3071
      const int row   = i / 192;
      const int chunk = i % 192;
      const uint32_t goff  = ((uint32_t)(bb * 16 + row) * T_ + (uint32_t)t) * (G3 * 4)
                           + (uint32_t)chunk * 16;
      const uint32_t laddr = ldsbase + (uint32_t)(row * (G3 * 4) + chunk * 16);
      asm volatile("global_load_async_to_lds_b128 %0, %1, %2"
                   :: "v"(laddr), "v"(goff), "s"(gi) : "memory");
    }
  };

  prefetch(0, lds0);

  for (int t = 0; t < T_; ++t) {
    const int buf = t & 1;
    const int tn  = (t + 1 < T_) ? t + 1 : t;
    prefetch(tn, buf ? lds0 : lds1);                    // next step into other buffer
    asm volatile("s_wait_asynccnt 6" ::: "memory");     // this step's 6 retired
    __syncthreads();                                    // + prev-step h writes visible

    const float* gb = gi_buf + buf * (16 * G3);

    // C init: r/u gates start from gi (both biases folded); candidate from b_hh.
    // gi_c is captured into registers here (before any wave can overwrite gb).
    v8f   acc[3];
    float gic[8];
#pragma unroll
    for (int r = 0; r < 8; ++r) {
      const int row = hiH ? r + 8 : r;
      acc[0][r] = gb[row * G3 + jc];
      acc[1][r] = bhh_c;
      acc[2][r] = gb[row * G3 + 2 * H_ + jc];
      gic[r]    = gb[row * G3 + H_ + jc];
    }

    // gates += h @ W_hh^T  (A from LDS h_bf, B stationary in registers)
#pragma unroll
    for (int kk = 0; kk < 8; ++kk) {
      const int abase = l16 * H_ + kk * 32 + hiH * 16;
      v16bf a = cat8(*(const v8bf*)(h_bf + abase),
                     *(const v8bf*)(h_bf + abase + 8));
#pragma unroll
      for (int g = 0; g < 3; ++g)
        acc[g] = __builtin_amdgcn_wmma_f32_16x16x32_bf16(
            false, a, false, Bf[g][kk], (short)0, acc[g], false, false);
    }
    __syncthreads();   // all A-reads of h done before anyone rewrites h

#pragma unroll
    for (int r = 0; r < 8; ++r) {
      const int row = hiH ? r + 8 : r;
      const float u  = sigmoid_(acc[2][r]);
      const float rr = sigmoid_(acc[0][r]);
      const float n  = tanh_(gic[r] + rr * acc[1][r]);
      const float ho = h_f[row * H_ + jc];
      const float hy = (1.0f - u) * ho + u * n;
      h_f[row * H_ + jc]  = hy;
      h_bf[row * H_ + jc] = (__bf16)hy;
      out[((size_t)(bb * 16 + row) * T_ + t) * H_ + jc] = hy;
    }
  }

  __syncthreads();
  for (int i = tid; i < 16 * H_; i += 512)
    hN[(size_t)(bb * 16 + (i >> 8)) * H_ + (i & 255)] = h_f[i];
}

// ---------------- launcher ----------------
extern "C" void kernel_launch(void* const* d_in, const int* in_sizes, int n_in,
                              void* d_out, int out_size, void* d_ws, size_t ws_size,
                              hipStream_t stream) {
  (void)in_sizes; (void)n_in; (void)out_size; (void)ws_size;
  const float* x   = (const float*)d_in[0];
  const float* h0  = (const float*)d_in[1];
  const float* Wih = (const float*)d_in[2];
  const float* bih = (const float*)d_in[3];
  const float* Whh = (const float*)d_in[4];
  const float* bhh = (const float*)d_in[5];
  float* out = (float*)d_out;
  float* hN  = out + (size_t)B_ * T_ * H_;

  char* ws = (char*)d_ws;
  __bf16* Wih_bf = (__bf16*)ws;                 // 384 KB
  __bf16* Whh_bf = (__bf16*)(ws + 393216);      // 384 KB
  float*  gi     = (float*)(ws + (1u << 20));   // 402.7 MB, 1MB-aligned

  gru_cvt_w<<<G3 * I_ / 256, 256, 0, stream>>>(Wih, Whh, Wih_bf, Whh_bf);
  gru_gi_gemm<<<B_ * T_ / 16, 256, 0, stream>>>(x, Wih_bf, bih, bhh, gi);

  const int shmem = 2 * 16 * G3 * 4 + 16 * H_ * 2 + 16 * H_ * 4;  // 122880 B
  hipFuncSetAttribute((const void*)gru_rnn,
                      hipFuncAttributeMaxDynamicSharedMemorySize, shmem);
  gru_rnn<<<4, 512, shmem, stream>>>(gi, h0, Whh_bf, bhh, out, hN);
}
